// IntraInterAtt_86921548137232
// MI455X (gfx1250) — compile-verified, matrix operand order checked by reading, not run on hardware
//
#include <hip/hip_runtime.h>
#include <hip/hip_bf16.h>
#include <math.h>

// Problem constants (match reference setup_inputs)
#define NN   2048      // nodes
#define DF   128       // feature dim
#define HH   64        // LSTM hidden
#define TKC  32        // WMMA K chunk (bf16 16x16x32)

typedef __attribute__((ext_vector_type(16))) __bf16 v16bf;
typedef __attribute__((ext_vector_type(8)))  float  v8f;

__device__ __forceinline__ __bf16 f2bf(float f) { return (__bf16)f; }

__device__ __forceinline__ float sigf(float x) { return 1.f / (1.f + __expf(-x)); }

// order-preserving float<->uint for atomicMax-based segment max
__device__ __forceinline__ unsigned f2key(float f) {
  unsigned u = __builtin_bit_cast(unsigned, f);
  return (u & 0x80000000u) ? ~u : (u | 0x80000000u);
}
__device__ __forceinline__ float key2f(unsigned k) {
  unsigned u = (k & 0x80000000u) ? (k & 0x7fffffffu) : ~k;
  return __builtin_bit_cast(float, u);
}

// Generic -> LDS byte offset (LDS aperture: low 32 bits = in-LDS address)
__device__ __forceinline__ unsigned lds_off(const void* p) {
  return (unsigned)(unsigned long long)p;
}
// CDNA5 async copy: 16B global -> LDS, tracked by ASYNCcnt, bypasses VGPRs.
__device__ __forceinline__ void async_g2l_b128(unsigned ldsaddr, const void* gaddr) {
  asm volatile("global_load_async_to_lds_b128 %0, %1, off"
               :: "v"(ldsaddr), "v"(gaddr) : "memory");
}
__device__ __forceinline__ void wait_asynccnt0() {
  asm volatile("s_wait_asynccnt 0" ::: "memory");
}

// load 16 consecutive fp32 from LDS and convert to a bf16 WMMA fragment
__device__ __forceinline__ v16bf frag16(const float* p) {
  v16bf r;
#pragma unroll
  for (int i = 0; i < 16; ++i) r[i] = (__bf16)p[i];
  return r;
}

// ---------------------------------------------------------------------------
// Generic GEMM: C[M,Nout] = A[M,K] * W[Nout,K]^T (+ bias), fp32 in/out,
// bf16 WMMA compute with f32 accumulate. M, Nout multiples of 128.
// WG = 256 threads = 8 waves; WG tile 128x128; wave tile 32x64 (2x4 WMMA).
// K chunked by 32 with LDS double buffering. Full chunks are staged fp32 with
// GLOBAL_LOAD_ASYNC_TO_LDS_B128 (no VGPR round trip, no guards); the single
// K-tail chunk uses a guarded scalar path. fp32->bf16 happens at fragment
// load time (VALU cvt co-executes with the XDL WMMA pipe).
// ---------------------------------------------------------------------------
__global__ __launch_bounds__(256)
void gemm_wmma_bias(const float* __restrict__ A, const float* __restrict__ W,
                    const float* __restrict__ bias, float* __restrict__ C,
                    int M, int Nout, int K) {
  __shared__ __align__(16) float As[2][128 * TKC];
  __shared__ __align__(16) float Bs[2][128 * TKC];
  const int tid  = threadIdx.x;
  const int lane = tid & 31;
  const int wave = tid >> 5;
  const int wm   = (wave & 3) * 32;     // wave row offset in tile
  const int wn   = (wave >> 2) * 64;    // wave col offset in tile
  const int m0   = blockIdx.x * 128;
  const int n0   = blockIdx.y * 128;

  v8f acc[2][4] = {};

  // full-chunk stage: 128 rows x 32 floats = 1024 x 16B segments per matrix;
  // 256 threads issue 4+4 async b128 copies each.
  auto load_async = [&](int buf, int k0) {
    const unsigned abase = lds_off(&As[buf][0]);
    const unsigned bbase = lds_off(&Bs[buf][0]);
#pragma unroll
    for (int i = 0; i < 4; ++i) {
      int e = tid + i * 256;          // 0..1023
      int r = e >> 3, s = (e & 7) * 4;
      unsigned loff = (unsigned)((r * TKC + s) * 4);
      async_g2l_b128(abase + loff, A + (size_t)(m0 + r) * K + k0 + s);
      async_g2l_b128(bbase + loff, W + (size_t)(n0 + r) * K + k0 + s);
    }
  };
  // K-tail stage (zero-padded, executed at most once per kernel)
  auto load_guard = [&](int buf, int k0) {
#pragma unroll
    for (int i = 0; i < 16; ++i) {
      int e = tid + i * 256;
      int r = e >> 5, c = e & 31;
      int k = k0 + c;
      As[buf][r * TKC + c] = (k < K) ? A[(size_t)(m0 + r) * K + k] : 0.f;
      Bs[buf][r * TKC + c] = (k < K) ? W[(size_t)(n0 + r) * K + k] : 0.f;
    }
  };
  auto load = [&](int buf, int k0) {
    if (k0 + TKC <= K) load_async(buf, k0);
    else load_guard(buf, k0);
  };

  const int nk = (K + TKC - 1) / TKC;
  load(0, 0);
  wait_asynccnt0();
  __syncthreads();

  const int rr = lane & 15;
  const int kh = (lane >> 4) * 16;

  for (int t = 0; t < nk; ++t) {
    const int cur = t & 1;
    if (t + 1 < nk) load(cur ^ 1, (t + 1) * TKC);
    if (t + 2 < nk) {  // L2 prefetch of the chunk after next (global_prefetch_b8)
      __builtin_prefetch(&A[(size_t)(m0 + (tid >> 1)) * K + (t + 2) * TKC + (tid & 1) * 16], 0, 1);
      __builtin_prefetch(&W[(size_t)(n0 + (tid >> 1)) * K + (t + 2) * TKC + (tid & 1) * 16], 0, 1);
    }
    v16bf af[2], bf[4];
#pragma unroll
    for (int i = 0; i < 2; ++i)
      af[i] = frag16(&As[cur][(wm + i * 16 + rr) * TKC + kh]);
#pragma unroll
    for (int j = 0; j < 4; ++j)
      bf[j] = frag16(&Bs[cur][(wn + j * 16 + rr) * TKC + kh]);
#pragma unroll
    for (int i = 0; i < 2; ++i)
#pragma unroll
      for (int j = 0; j < 4; ++j)
        acc[i][j] = __builtin_amdgcn_wmma_f32_16x16x32_bf16(
            false, af[i], false, bf[j], (short)0, acc[i][j], false, false);
    wait_asynccnt0();   // async writes into the next buffer are LDS-visible
    __syncthreads();
  }

  // C/D layout: VGPR r -> lanes 0-15: M=r, N=lane; lanes 16-31: M=r+8, N=lane-16
  const int cn   = lane & 15;
  const int roff = (lane >> 4) * 8;
#pragma unroll
  for (int i = 0; i < 2; ++i)
#pragma unroll
    for (int j = 0; j < 4; ++j)
#pragma unroll
      for (int r = 0; r < 8; ++r) {
        int row  = m0 + wm + i * 16 + roff + r;
        int colc = n0 + wn + j * 16 + cn;
        float v = acc[i][j][r];
        if (bias) v += bias[colc];
        C[(size_t)row * Nout + colc] = v;
      }
}

// ---------------------------------------------------------------------------
// Fused LSTM step: gates = [X_t | H] @ [Wih | Whh]^T + (bih+bhh), then the
// i/f/g/o nonlinearities and c/h update, writing h into the (N,5,128) output.
// One launch per (direction, step); grid = 16 WGs of 128 rows. Each of the 8
// waves computes a 16-row x 256-col gate tile (16 WMMA accumulators), so each
// lane owns all four gates of its hidden unit -> fully fused epilogue.
// ---------------------------------------------------------------------------
__global__ __launch_bounds__(256)
void lstm_step_kernel(const float* __restrict__ xbuf,
                      const float* __restrict__ wih, const float* __restrict__ whh,
                      const float* __restrict__ bih, const float* __restrict__ bhh,
                      float* __restrict__ h_state, float* __restrict__ c_state,
                      float* __restrict__ hout, int step, int dir) {
  __shared__ __align__(16) __bf16 As[128 * TKC];   // [X_t | H] chunk: 128 x 32
  __shared__ __align__(16) __bf16 Bs[256 * TKC];   // [Wih | Whh] chunk: 256 x 32
  __shared__ float bb[256];

  const int tid  = threadIdx.x;
  const int lane = tid & 31;
  const int wave = tid >> 5;
  const int base = blockIdx.x * 128;
  const int t_io = dir ? (4 - step) : step;  // same index for input & output

  bb[tid] = bih[tid] + bhh[tid];

  v8f acc[16] = {};
  const int rr = lane & 15;
  const int kh = (lane >> 4) * 16;

  for (int t = 0; t < 6; ++t) {  // K = 192 = 6 * 32
    __syncthreads();
    const int k0 = t * TKC;
#pragma unroll
    for (int i = 0; i < 16; ++i) {  // A chunk: 4096 elems
      int e = tid + i * 256;
      int r = e >> 5, c = e & 31;
      int k = k0 + c;
      float v;
      if (k < 128)      v = xbuf[(size_t)(base + r) * 640 + t_io * 128 + k];
      else if (step == 0) v = 0.f;
      else              v = h_state[(base + r) * HH + (k - 128)];
      As[r * TKC + c] = f2bf(v);
    }
#pragma unroll
    for (int i = 0; i < 32; ++i) {  // B chunk: 8192 elems
      int e = tid + i * 256;
      int r = e >> 5, c = e & 31;
      int k = k0 + c;
      float v = (k < 128) ? wih[r * 128 + k] : whh[r * HH + (k - 128)];
      Bs[r * TKC + c] = f2bf(v);
    }
    __syncthreads();
    v16bf af = *(const v16bf*)(As + (wave * 16 + rr) * TKC + kh);
#pragma unroll
    for (int j = 0; j < 16; ++j) {
      v16bf bf = *(const v16bf*)(Bs + (j * 16 + rr) * TKC + kh);
      acc[j] = __builtin_amdgcn_wmma_f32_16x16x32_bf16(
          false, af, false, bf, (short)0, acc[j], false, false);
    }
  }

  const int cn   = lane & 15;
  const int roff = (lane >> 4) * 8;
#pragma unroll
  for (int jt = 0; jt < 4; ++jt) {
#pragma unroll
    for (int r = 0; r < 8; ++r) {
      int node = base + wave * 16 + roff + r;
      int hc   = jt * 16 + cn;                 // hidden index in [0,64)
      float ig = sigf(acc[jt][r]      + bb[hc]);
      float fg = sigf(acc[jt + 4][r]  + bb[64 + hc]);
      float gg = tanhf(acc[jt + 8][r] + bb[128 + hc]);
      float og = sigf(acc[jt + 12][r] + bb[192 + hc]);
      float co = (step == 0) ? 0.f : c_state[node * HH + hc];
      float cnew = fg * co + ig * gg;
      float hnew = og * tanhf(cnew);
      c_state[node * HH + hc] = cnew;
      h_state[node * HH + hc] = hnew;
      hout[(size_t)node * 640 + t_io * 128 + dir * HH + hc] = hnew;
    }
  }
}

// ---------------------------- small utility kernels ------------------------
__global__ void fill_f32(float* p, float v, int n) {
  int i = blockIdx.x * blockDim.x + threadIdx.x;
  if (i < n) p[i] = v;
}
__global__ void fill_u32(unsigned* p, unsigned v, int n) {
  int i = blockIdx.x * blockDim.x + threadIdx.x;
  if (i < n) p[i] = v;
}
__global__ void add_vec(const float* a, const float* b, float* o, int n) {
  int i = blockIdx.x * blockDim.x + threadIdx.x;
  if (i < n) o[i] = a[i] + b[i];
}
__global__ void mean5_kernel(const float* __restrict__ hout, float* __restrict__ xm) {
  int i = blockIdx.x * blockDim.x + threadIdx.x;
  if (i >= NN * DF) return;
  int n = i >> 7, d = i & 127;
  float s = 0.f;
#pragma unroll
  for (int t = 0; t < 5; ++t) s += hout[(size_t)n * 640 + t * 128 + d];
  xm[i] = s * 0.2f;
}

// ---- inter-graph GCN (sparse, atomic scatter) ----
__global__ void edge_deg_kernel(const int* __restrict__ col, const float* __restrict__ ew,
                                float* __restrict__ deg, int E) {
  int e = blockIdx.x * blockDim.x + threadIdx.x;
  if (e < E) unsafeAtomicAdd(&deg[col[e]], ew[e]);
}
__global__ void dinv_kernel(float* deg, int n) {
  int i = blockIdx.x * blockDim.x + threadIdx.x;
  if (i < n) deg[i] = rsqrtf(deg[i] + 1.f);  // +1 self-loop; always > 0
}
__global__ void gcn_self_kernel(const float* __restrict__ xw, const float* __restrict__ dinv,
                                const float* __restrict__ b, float* __restrict__ out) {
  int i = blockIdx.x * blockDim.x + threadIdx.x;
  if (i >= NN * DF) return;
  int n = i >> 7, d = i & 127;
  float di = dinv[n];
  out[i] = b[d] + di * di * xw[i];
}
__global__ void gcn_scatter_kernel(const int* __restrict__ row, const int* __restrict__ col,
                                   const float* __restrict__ ew, const float* __restrict__ dinv,
                                   const float* __restrict__ xw, float* __restrict__ out, int E) {
  int gw = (blockIdx.x * blockDim.x + threadIdx.x) >> 5;
  int lane = threadIdx.x & 31;
  if (gw >= E) return;
  int r = row[gw], c = col[gw];
  float nrm = dinv[r] * ew[gw] * dinv[c];
#pragma unroll
  for (int q = 0; q < 4; ++q) {
    int d = lane + 32 * q;
    unsafeAtomicAdd(&out[(size_t)c * DF + d], nrm * xw[(size_t)r * DF + d]);
  }
}

// ---- inter-graph TransformerConv (sparse softmax attention) ----
__global__ void edge_alpha_kernel(const int* __restrict__ row, const int* __restrict__ col,
                                  const float* __restrict__ q, const float* __restrict__ k,
                                  float* __restrict__ alpha, unsigned* __restrict__ mkey, int E) {
  int gw = (blockIdx.x * blockDim.x + threadIdx.x) >> 5;
  int lane = threadIdx.x & 31;
  if (gw >= E) return;
  int s = row[gw], d = col[gw];
  float p = 0.f;
#pragma unroll
  for (int qq = 0; qq < 4; ++qq) {
    int dd = lane + 32 * qq;
    p += q[(size_t)d * DF + dd] * k[(size_t)s * DF + dd];
  }
#pragma unroll
  for (int off = 16; off; off >>= 1) p += __shfl_down(p, off, 32);
  if (lane == 0) {
    p *= 0.08838834764831845f;  // 1/sqrt(128)
    alpha[gw] = p;
    atomicMax(&mkey[d], f2key(p));
  }
}
__global__ void edge_expsum_kernel(const int* __restrict__ col, const float* __restrict__ alpha,
                                   const unsigned* __restrict__ mkey, float* __restrict__ ealpha,
                                   float* __restrict__ ssum, int E) {
  int e = blockIdx.x * blockDim.x + threadIdx.x;
  if (e >= E) return;
  int d = col[e];
  float ev = __expf(alpha[e] - key2f(mkey[d]));
  ealpha[e] = ev;
  unsafeAtomicAdd(&ssum[d], ev);
}
__global__ void attn_scatter_kernel(const int* __restrict__ row, const int* __restrict__ col,
                                    const float* __restrict__ ealpha, const float* __restrict__ ssum,
                                    const float* __restrict__ v, float* __restrict__ out, int E) {
  int gw = (blockIdx.x * blockDim.x + threadIdx.x) >> 5;
  int lane = threadIdx.x & 31;
  if (gw >= E) return;
  int s = row[gw], d = col[gw];
  float a = ealpha[gw] / ssum[d];
#pragma unroll
  for (int q = 0; q < 4; ++q) {
    int dd = lane + 32 * q;
    unsafeAtomicAdd(&out[(size_t)d * DF + dd], a * v[(size_t)s * DF + dd]);
  }
}

// ---- final heads: out1 = x@fc^T+b ; out2 = x@fcm^T+bm ----
__global__ void final_fc_kernel(const float* __restrict__ x,
                                const float* __restrict__ fw, const float* __restrict__ fb,
                                const float* __restrict__ fmw, const float* __restrict__ fmb,
                                float* __restrict__ out) {
  int gw = (blockIdx.x * blockDim.x + threadIdx.x) >> 5;
  int lane = threadIdx.x & 31;
  if (gw >= NN) return;
  float a = 0.f, b = 0.f;
#pragma unroll
  for (int q = 0; q < 4; ++q) {
    int d = lane + 32 * q;
    float xv = x[(size_t)gw * DF + d];
    a += xv * fw[d];
    b += xv * fmw[d];
  }
#pragma unroll
  for (int off = 16; off; off >>= 1) {
    a += __shfl_down(a, off, 32);
    b += __shfl_down(b, off, 32);
  }
  if (lane == 0) {
    out[gw]      = a + fb[0];
    out[NN + gw] = b + fmb[0];
  }
}

// Param order = params dict insertion order, starting at d_in[9]
enum {
  ATT1_W = 0, ATT1_B, ATT2_W, ATT2_B, ATT3_W, ATT3_B, ATT4_W, ATT4_B,
  ATT5A_W, ATT5A_B, ATT5B_W, ATT5B_B,
  LF_WIH, LF_WHH, LF_BIH, LF_BHH, LB_WIH, LB_WHH, LB_BIH, LB_BHH,
  GCN1_W, GCN1_B, IT_Q_W, IT_Q_B, IT_K_W, IT_K_B, IT_V_W, IT_V_B, IT_S_W, IT_S_B,
  GCN2_W, GCN2_B, ET_Q_W, ET_Q_B, ET_K_W, ET_K_B, ET_V_W, ET_V_B, ET_S_W, ET_S_B,
  FC_W, FC_B, FCM_W, FCM_B
};

extern "C" void kernel_launch(void* const* d_in, const int* in_sizes, int n_in,
                              void* d_out, int out_size, void* d_ws, size_t ws_size,
                              hipStream_t stream) {
  (void)n_in; (void)out_size; (void)ws_size;
  const float* x1 = (const float*)d_in[0];
  const float* x2 = (const float*)d_in[1];
  const float* x3 = (const float*)d_in[2];
  const float* x4 = (const float*)d_in[3];
  const float* x5 = (const float*)d_in[4];
  const int* eidx = (const int*)d_in[5];
  const float* ewg = (const float*)d_in[6];
  const int E  = in_sizes[5] / 2;        // 32768
  const int K5 = in_sizes[4] / NN;       // 96961
  auto prm = [&](int i) { return (const float*)d_in[9 + i]; };

  // workspace carve-out
  char* wsb = (char*)d_ws;
  size_t off = 0;
  auto alloc = [&](size_t nbytes) {
    void* p = wsb + off;
    off += (nbytes + 255) & ~(size_t)255;
    return p;
  };
  float* xbuf  = (float*)alloc((size_t)NN * 5 * DF * 4);  // stacked h1..h5 == reshaped (N,5,D)
  float* t5    = (float*)alloc((size_t)NN * 1024 * 4);
  float* hout  = (float*)alloc((size_t)NN * 5 * DF * 4);  // LSTM output (N,5,128)
  float* hst   = (float*)alloc((size_t)2 * NN * HH * 4);
  float* cst   = (float*)alloc((size_t)2 * NN * HH * 4);
  float* xm    = (float*)alloc((size_t)NN * DF * 4);
  float* gbuf  = (float*)alloc((size_t)NN * DF * 4);
  float* wc    = (float*)alloc((size_t)DF * DF * 4);
  float* bc    = (float*)alloc((size_t)DF * 4);
  float* intra = (float*)alloc((size_t)NN * DF * 4);
  float* xw    = (float*)alloc((size_t)NN * DF * 4);
  float* deg   = (float*)alloc((size_t)NN * 4);
  float* inter1= (float*)alloc((size_t)NN * DF * 4);
  float* qb    = (float*)alloc((size_t)NN * DF * 4);
  float* kb    = (float*)alloc((size_t)NN * DF * 4);
  float* vb    = (float*)alloc((size_t)NN * DF * 4);
  float* tout  = (float*)alloc((size_t)NN * DF * 4);
  float* alpha = (float*)alloc((size_t)E * 4);
  float* ealpha= (float*)alloc((size_t)E * 4);
  unsigned* mkey = (unsigned*)alloc((size_t)NN * 4);
  float* ssum  = (float*)alloc((size_t)NN * 4);

  const dim3 B256(256);

  // --- attribute embeddings; stacked-reshape is just contiguous placement ---
  gemm_wmma_bias<<<dim3(16, 1), B256, 0, stream>>>(x1, prm(ATT1_W), prm(ATT1_B), xbuf + 0 * NN * DF, NN, DF, 17);
  gemm_wmma_bias<<<dim3(16, 1), B256, 0, stream>>>(x2, prm(ATT2_W), prm(ATT2_B), xbuf + 1 * NN * DF, NN, DF, 17);
  gemm_wmma_bias<<<dim3(16, 1), B256, 0, stream>>>(x3, prm(ATT3_W), prm(ATT3_B), xbuf + 2 * NN * DF, NN, DF, 6);
  gemm_wmma_bias<<<dim3(16, 1), B256, 0, stream>>>(x4, prm(ATT4_W), prm(ATT4_B), xbuf + 3 * NN * DF, NN, DF, 6);
  // dominant GEMM: 2048 x 96961 x 1024
  gemm_wmma_bias<<<dim3(16, 8), B256, 0, stream>>>(x5, prm(ATT5A_W), prm(ATT5A_B), t5, NN, 1024, K5);
  gemm_wmma_bias<<<dim3(16, 1), B256, 0, stream>>>(t5, prm(ATT5B_W), prm(ATT5B_B), xbuf + 4 * NN * DF, NN, DF, 1024);

  // --- bidirectional LSTM over T=5 (10 fused WMMA+activation launches) ---
  for (int s = 0; s < 5; ++s) {
    lstm_step_kernel<<<16, B256, 0, stream>>>(xbuf, prm(LF_WIH), prm(LF_WHH), prm(LF_BIH), prm(LF_BHH),
                                              hst, cst, hout, s, 0);
    lstm_step_kernel<<<16, B256, 0, stream>>>(xbuf, prm(LB_WIH), prm(LB_WHH), prm(LB_BIH), prm(LB_BHH),
                                              hst + NN * HH, cst + NN * HH, hout, s, 1);
  }

  // --- intra GCN + intra transformer (exact: uniform softmax over identical rows) ---
  mean5_kernel<<<(NN * DF) / 256, B256, 0, stream>>>(hout, xm);
  gemm_wmma_bias<<<dim3(16, 1), B256, 0, stream>>>(xm, prm(GCN1_W), prm(GCN1_B), gbuf, NN, DF, DF);
  add_vec<<<64, B256, 0, stream>>>(prm(IT_V_W), prm(IT_S_W), wc, DF * DF);
  add_vec<<<1, B256, 0, stream>>>(prm(IT_V_B), prm(IT_S_B), bc, DF);
  gemm_wmma_bias<<<dim3(16, 1), B256, 0, stream>>>(gbuf, wc, bc, intra, NN, DF, DF);

  // --- inter sparse GCN ---
  gemm_wmma_bias<<<dim3(16, 1), B256, 0, stream>>>(intra, prm(GCN2_W), nullptr, xw, NN, DF, DF);
  fill_f32<<<(NN + 255) / 256, B256, 0, stream>>>(deg, 0.f, NN);
  edge_deg_kernel<<<(E + 255) / 256, B256, 0, stream>>>(eidx + E, ewg, deg, E);
  dinv_kernel<<<(NN + 255) / 256, B256, 0, stream>>>(deg, NN);
  gcn_self_kernel<<<(NN * DF) / 256, B256, 0, stream>>>(xw, deg, prm(GCN2_B), inter1);
  gcn_scatter_kernel<<<(E * 32) / 256, B256, 0, stream>>>(eidx, eidx + E, ewg, deg, xw, inter1, E);

  // --- inter sparse TransformerConv ---
  gemm_wmma_bias<<<dim3(16, 1), B256, 0, stream>>>(inter1, prm(ET_Q_W), prm(ET_Q_B), qb, NN, DF, DF);
  gemm_wmma_bias<<<dim3(16, 1), B256, 0, stream>>>(inter1, prm(ET_K_W), prm(ET_K_B), kb, NN, DF, DF);
  gemm_wmma_bias<<<dim3(16, 1), B256, 0, stream>>>(inter1, prm(ET_V_W), prm(ET_V_B), vb, NN, DF, DF);
  gemm_wmma_bias<<<dim3(16, 1), B256, 0, stream>>>(inter1, prm(ET_S_W), prm(ET_S_B), tout, NN, DF, DF);
  fill_u32<<<(NN + 255) / 256, B256, 0, stream>>>(mkey, 0x007FFFFFu, NN);  // key(-inf)
  fill_f32<<<(NN + 255) / 256, B256, 0, stream>>>(ssum, 0.f, NN);
  edge_alpha_kernel<<<(E * 32) / 256, B256, 0, stream>>>(eidx, eidx + E, qb, kb, alpha, mkey, E);
  edge_expsum_kernel<<<(E + 255) / 256, B256, 0, stream>>>(eidx + E, alpha, mkey, ealpha, ssum, E);
  attn_scatter_kernel<<<(E * 32) / 256, B256, 0, stream>>>(eidx, eidx + E, ealpha, ssum, vb, tout, E);

  // --- output heads (out1 || out2 flat) ---
  final_fc_kernel<<<(NN * 32) / 256, B256, 0, stream>>>(tout, prm(FC_W), prm(FC_B),
                                                        prm(FCM_W), prm(FCM_B), (float*)d_out);
}
